// TFBranch_33423435497779
// MI455X (gfx1250) — compile-verified, matrix operand order checked by reading, not run on hardware
//
#include <hip/hip_runtime.h>
#include <math.h>

// ---------------------------------------------------------------------------
// Fused CfC network for MI455X (gfx1250, wave32, WMMA).
// Each wave processes a 16-row batch tile end-to-end:
//   GEMM1(116->64) -> LN -> GELU -> cell0(84->20) -> cell1(32->12)
//   -> cell2(76->64) -> LN.
// Matmuls use v_wmma_f32_16x16x32_bf16 (bf16 in, f32 accumulate).
// ta/tb matrices are pre-summed (sigmoid(a+b) identity) -> 3 matmuls/cell.
// Weights live in LDS as bf16 with odd-word strides (conflict-free gathers).
// ---------------------------------------------------------------------------

typedef __attribute__((ext_vector_type(16))) __bf16 v16bf;
typedef __attribute__((ext_vector_type(8)))  float  v8f;

union ABFrag { v16bf v; unsigned int u[8]; };
union CDFrag { v8f  v; float        f[8]; };

#define MEMFENCE() asm volatile("" ::: "memory")

__device__ __forceinline__ unsigned short f2bf(float f) {
  unsigned int u = __float_as_uint(f);
  u += 0x7fffu + ((u >> 16) & 1u);   // round-to-nearest-even
  return (unsigned short)(u >> 16);
}

// Build a 16x32 bf16 fragment (A layout; B uses same layout = column-major).
// lane: m (or n) = lane&15 passed as `row`; half = lane>>4.
// VGPR v holds K pair index (v>>2)*8 + half*4 + (v&3)  (+ kp_base).
__device__ __forceinline__ v16bf make_frag(const unsigned int* __restrict__ base,
                                           int row, int stride32, int kp_base, int half) {
  ABFrag a;
  const unsigned int* p = base + row * stride32 + kp_base + half * 4;
#pragma unroll
  for (int v = 0; v < 8; ++v)
    a.u[v] = p[(v >> 2) * 8 + (v & 3)];
  return a.v;
}

__device__ __forceinline__ v8f wmma_bf16(v16bf a, v16bf b, v8f c) {
  return __builtin_amdgcn_wmma_f32_16x16x32_bf16(false, a, false, b, (short)0, c,
                                                 false, false);
}

__device__ __forceinline__ float geluf(float x) {
  return 0.5f * x * (1.0f + erff(x * 0.70710678118654752f));
}
__device__ __forceinline__ float sigmf(float x) {
  return 1.0f / (1.0f + expf(-x));
}

__device__ __forceinline__ void cvt_mat(const float* __restrict__ g,
                                        unsigned short* __restrict__ l,
                                        int rows, int cols, int stride,
                                        int tid, int nthr) {
  int n = rows * cols;
  for (int i = tid; i < n; i += nthr) {
    int r = i / cols, c = i - r * cols;
    l[r * stride + c] = f2bf(g[i]);
  }
}
__device__ __forceinline__ void cvt_mat2(const float* __restrict__ ga,
                                         const float* __restrict__ gb,
                                         unsigned short* __restrict__ l,
                                         int rows, int cols, int stride,
                                         int tid, int nthr) {
  int n = rows * cols;
  for (int i = tid; i < n; i += nthr) {
    int r = i / cols, c = i - r * cols;
    l[r * stride + c] = f2bf(ga[i] + gb[i]);
  }
}

// ---- LDS layout (bytes) ----
#define SM_WIN   0                    // in_w  bf16 [64][130]          16640
#define SM_W0    16640                // 3x [32][98] bf16 (f1,f2,ta+tb) 18816
#define SM_W0SZ  6272
#define SM_W1    35456                // 3x [16][34] bf16                3264
#define SM_W1SZ  1088
#define SM_W2    38720                // 3x [64][98] bf16               37632
#define SM_W2SZ  12544
#define SM_BIAS  76352                // 656 f32                         2624
#define SM_WAVE  78976                // per-wave region base
#define PW       15680                // per-wave bytes
#define PW_CATIN 0                    // [16][130] bf16  4160
#define PW_CAT0  4160                 // [16][98]  bf16  3136
#define PW_CAT1  7296                 // [16][34]  bf16  1088
#define PW_CAT2  8384                 // [16][98]  bf16  3136
#define PW_LNS   11520                // [16][65]  f32   4160
#define SMEM_TOTAL (SM_WAVE + 4 * PW) // 141,696 B

extern "C" __global__ void __launch_bounds__(128, 1)
cfc_fused_kernel(const float* __restrict__ per_tf, const float* __restrict__ shared_f,
                 const float* __restrict__ hx,
                 const float* __restrict__ in_w, const float* __restrict__ in_b,
                 const float* __restrict__ ln1g, const float* __restrict__ ln1b,
                 const float* __restrict__ ln2g, const float* __restrict__ ln2b,
                 const float* __restrict__ f1w0, const float* __restrict__ f1b0,
                 const float* __restrict__ f2w0, const float* __restrict__ f2b0,
                 const float* __restrict__ taw0, const float* __restrict__ tab0,
                 const float* __restrict__ tbw0, const float* __restrict__ tbb0,
                 const float* __restrict__ f1w1, const float* __restrict__ f1b1,
                 const float* __restrict__ f2w1, const float* __restrict__ f2b1,
                 const float* __restrict__ taw1, const float* __restrict__ tab1,
                 const float* __restrict__ tbw1, const float* __restrict__ tbb1,
                 const float* __restrict__ f1w2, const float* __restrict__ f1b2,
                 const float* __restrict__ f2w2, const float* __restrict__ f2b2,
                 const float* __restrict__ taw2, const float* __restrict__ tab2,
                 const float* __restrict__ tbw2, const float* __restrict__ tbb2,
                 float* __restrict__ out, float* __restrict__ hx_out, int ntiles) {
  __shared__ __align__(16) unsigned char smem[SMEM_TOTAL];

  const int tid  = threadIdx.x;
  const int lane = tid & 31;
  const int wave = tid >> 5;
  const int hf   = lane >> 4;   // K-half selector
  const int nl   = lane & 15;   // m (A) or n (B/C/D) within 16-tile

  // ---- zero all of LDS (covers every zero-padded region) ----
  {
    unsigned int* p = (unsigned int*)smem;
    for (int i = tid; i < SMEM_TOTAL / 4; i += 128) p[i] = 0u;
  }
  __syncthreads();

  // ---- weight / bias staging ----
  unsigned short* Win  = (unsigned short*)(smem + SM_WIN);
  unsigned short* W0f1 = (unsigned short*)(smem + SM_W0);
  unsigned short* W0f2 = (unsigned short*)(smem + SM_W0 + SM_W0SZ);
  unsigned short* W0t  = (unsigned short*)(smem + SM_W0 + 2 * SM_W0SZ);
  unsigned short* W1f1 = (unsigned short*)(smem + SM_W1);
  unsigned short* W1f2 = (unsigned short*)(smem + SM_W1 + SM_W1SZ);
  unsigned short* W1t  = (unsigned short*)(smem + SM_W1 + 2 * SM_W1SZ);
  unsigned short* W2f1 = (unsigned short*)(smem + SM_W2);
  unsigned short* W2f2 = (unsigned short*)(smem + SM_W2 + SM_W2SZ);
  unsigned short* W2t  = (unsigned short*)(smem + SM_W2 + 2 * SM_W2SZ);

  float* bias = (float*)(smem + SM_BIAS);
  float* Bin  = bias;          float* G1 = bias + 64;  float* B1v = bias + 128;
  float* G2   = bias + 192;    float* B2v = bias + 256;
  float* b0f1 = bias + 320;    float* b0f2 = bias + 352; float* b0t = bias + 384; // 32 each
  float* b1f1 = bias + 416;    float* b1f2 = bias + 432; float* b1t = bias + 448; // 16 each
  float* b2f1 = bias + 464;    float* b2f2 = bias + 528; float* b2t = bias + 592; // 64 each

  cvt_mat (in_w,        Win,  64, 116, 130, tid, 128);
  cvt_mat (f1w0,        W0f1, 20,  84,  98, tid, 128);
  cvt_mat (f2w0,        W0f2, 20,  84,  98, tid, 128);
  cvt_mat2(taw0, tbw0,  W0t,  20,  84,  98, tid, 128);
  cvt_mat (f1w1,        W1f1, 12,  32,  34, tid, 128);
  cvt_mat (f2w1,        W1f2, 12,  32,  34, tid, 128);
  cvt_mat2(taw1, tbw1,  W1t,  12,  32,  34, tid, 128);
  cvt_mat (f1w2,        W2f1, 64,  76,  98, tid, 128);
  cvt_mat (f2w2,        W2f2, 64,  76,  98, tid, 128);
  cvt_mat2(taw2, tbw2,  W2t,  64,  76,  98, tid, 128);

  for (int i = tid; i < 64; i += 128) {
    Bin[i] = in_b[i];  G1[i] = ln1g[i];  B1v[i] = ln1b[i];
    G2[i] = ln2g[i];   B2v[i] = ln2b[i];
    b2f1[i] = f1b2[i]; b2f2[i] = f2b2[i]; b2t[i] = tab2[i] + tbb2[i];
  }
  for (int i = tid; i < 20; i += 128) {
    b0f1[i] = f1b0[i]; b0f2[i] = f2b0[i]; b0t[i] = tab0[i] + tbb0[i];
  }
  for (int i = tid; i < 12; i += 128) {
    b1f1[i] = f1b1[i]; b1f2[i] = f2b1[i]; b1t[i] = tab1[i] + tbb1[i];
  }
  __syncthreads();

  // ---- per-wave activation buffers ----
  unsigned char* pw = smem + SM_WAVE + wave * PW;
  unsigned short* catIn = (unsigned short*)(pw + PW_CATIN); // [16][130]
  unsigned short* cat0  = (unsigned short*)(pw + PW_CAT0);  // [16][98]: 0-63 x, 64-83 h0, 84.. 0
  unsigned short* cat1  = (unsigned short*)(pw + PW_CAT1);  // [16][34]: 0-19 nh0, 20-31 h1
  unsigned short* cat2  = (unsigned short*)(pw + PW_CAT2);  // [16][98]: 0-11 nh1, 12-75 h2, 76.. 0
  float*          lns   = (float*)(pw + PW_LNS);            // [16][65]

  const int slot   = blockIdx.x * 4 + wave;
  const int nslots = gridDim.x * 4;

  for (int t = slot; t < ntiles; t += nslots) {
    const int r0 = t * 16;

    // ---- stage inputs into LDS (f32 -> bf16) ----
    for (int i = lane; i < 16 * 51; i += 32) {
      int r = i / 51, c = i - r * 51;
      catIn[r * 130 + c] = f2bf(per_tf[(size_t)(r0 + r) * 51 + c]);
    }
    for (int i = lane; i < 16 * 65; i += 32) {
      int r = i / 65, c = i - r * 65;
      catIn[r * 130 + 51 + c] = f2bf(shared_f[(size_t)(r0 + r) * 65 + c]);
    }
    for (int i = lane; i < 16 * 96; i += 32) {
      int r = i / 96, c = i - r * 96;
      unsigned short v = f2bf(hx[(size_t)(r0 + r) * 96 + c]);
      if (c < 20)      cat0[r * 98 + 64 + c] = v;
      else if (c < 32) cat1[r * 34 + 20 + (c - 20)] = v;
      else             cat2[r * 98 + 12 + (c - 32)] = v;
    }
    MEMFENCE();

    // ---- GEMM1: [16,116(pad128)] x [116,64] ----
    CDFrag acc[4];
#pragma unroll
    for (int n = 0; n < 4; ++n)
#pragma unroll
      for (int i = 0; i < 8; ++i) acc[n].f[i] = 0.0f;
#pragma unroll
    for (int kt = 0; kt < 4; ++kt) {
      v16bf a = make_frag((const unsigned int*)catIn, nl, 65, kt * 16, hf);
#pragma unroll
      for (int nt = 0; nt < 4; ++nt) {
        v16bf b = make_frag((const unsigned int*)Win, nt * 16 + nl, 65, kt * 16, hf);
        acc[nt].v = wmma_bf16(a, b, acc[nt].v);
      }
    }
#pragma unroll
    for (int nt = 0; nt < 4; ++nt) {
      float bv = Bin[nt * 16 + nl];
#pragma unroll
      for (int v = 0; v < 8; ++v)
        lns[(v + hf * 8) * 65 + nt * 16 + nl] = acc[nt].f[v] + bv;
    }
    MEMFENCE();

    // ---- LN1 + GELU -> cat0[:, 0:64] ----
    if (lane < 16) {
      float* rp = lns + lane * 65;
      float s = 0.0f, s2 = 0.0f;
      for (int c = 0; c < 64; ++c) { float x = rp[c]; s += x; s2 += x * x; }
      float mu = s * 0.015625f;
      float rs = rsqrtf(s2 * 0.015625f - mu * mu + 1e-5f);
      for (int c = 0; c < 64; ++c) {
        float y = (rp[c] - mu) * rs * G1[c] + B1v[c];
        cat0[lane * 98 + c] = f2bf(geluf(y));
      }
    }
    MEMFENCE();

    // ---- cell0: [16,84(pad96)] -> 20 units (2 N-tiles), 3 mats ----
    CDFrag c0a[2], c0b[2], c0c[2];
#pragma unroll
    for (int n = 0; n < 2; ++n)
#pragma unroll
      for (int i = 0; i < 8; ++i) { c0a[n].f[i] = 0.f; c0b[n].f[i] = 0.f; c0c[n].f[i] = 0.f; }
#pragma unroll
    for (int kt = 0; kt < 3; ++kt) {
      v16bf a = make_frag((const unsigned int*)cat0, nl, 49, kt * 16, hf);
#pragma unroll
      for (int nt = 0; nt < 2; ++nt) {
        c0a[nt].v = wmma_bf16(a, make_frag((const unsigned int*)W0f1, nt * 16 + nl, 49, kt * 16, hf), c0a[nt].v);
        c0b[nt].v = wmma_bf16(a, make_frag((const unsigned int*)W0f2, nt * 16 + nl, 49, kt * 16, hf), c0b[nt].v);
        c0c[nt].v = wmma_bf16(a, make_frag((const unsigned int*)W0t,  nt * 16 + nl, 49, kt * 16, hf), c0c[nt].v);
      }
    }
#pragma unroll
    for (int nt = 0; nt < 2; ++nt) {
      int n = nt * 16 + nl;
      float bv1 = b0f1[n], bv2 = b0f2[n], bvt = b0t[n];
#pragma unroll
      for (int v = 0; v < 8; ++v) {
        int m = v + hf * 8;
        float ff1 = tanhf(c0a[nt].f[v] + bv1);
        float ff2 = tanhf(c0b[nt].f[v] + bv2);
        float tt  = sigmf(c0c[nt].f[v] + bvt);
        float nh  = ff1 * (1.0f - tt) + tt * ff2;
        if (n < 20) {
          cat1[m * 34 + n] = f2bf(nh);
          hx_out[(size_t)(r0 + m) * 96 + n] = nh;
        }
      }
    }
    MEMFENCE();

    // ---- cell1: [16,32] -> 12 units (1 N-tile), 3 mats ----
    CDFrag c1a, c1b, c1c;
#pragma unroll
    for (int i = 0; i < 8; ++i) { c1a.f[i] = 0.f; c1b.f[i] = 0.f; c1c.f[i] = 0.f; }
    {
      v16bf a = make_frag((const unsigned int*)cat1, nl, 17, 0, hf);
      c1a.v = wmma_bf16(a, make_frag((const unsigned int*)W1f1, nl, 17, 0, hf), c1a.v);
      c1b.v = wmma_bf16(a, make_frag((const unsigned int*)W1f2, nl, 17, 0, hf), c1b.v);
      c1c.v = wmma_bf16(a, make_frag((const unsigned int*)W1t,  nl, 17, 0, hf), c1c.v);
    }
    {
      int n = nl;
      float bv1 = b1f1[n], bv2 = b1f2[n], bvt = b1t[n];
#pragma unroll
      for (int v = 0; v < 8; ++v) {
        int m = v + hf * 8;
        float ff1 = tanhf(c1a.f[v] + bv1);
        float ff2 = tanhf(c1b.f[v] + bv2);
        float tt  = sigmf(c1c.f[v] + bvt);
        float nh  = ff1 * (1.0f - tt) + tt * ff2;
        if (n < 12) {
          cat2[m * 98 + n] = f2bf(nh);
          hx_out[(size_t)(r0 + m) * 96 + 20 + n] = nh;
        }
      }
    }
    MEMFENCE();

    // ---- cell2: [16,76(pad96)] -> 64 units (4 N-tiles), 3 mats ----
    CDFrag d1[4], d2[4], d3[4];
#pragma unroll
    for (int n = 0; n < 4; ++n)
#pragma unroll
      for (int i = 0; i < 8; ++i) { d1[n].f[i] = 0.f; d2[n].f[i] = 0.f; d3[n].f[i] = 0.f; }
#pragma unroll
    for (int kt = 0; kt < 3; ++kt) {
      v16bf a = make_frag((const unsigned int*)cat2, nl, 49, kt * 16, hf);
#pragma unroll
      for (int nt = 0; nt < 4; ++nt) {
        d1[nt].v = wmma_bf16(a, make_frag((const unsigned int*)W2f1, nt * 16 + nl, 49, kt * 16, hf), d1[nt].v);
        d2[nt].v = wmma_bf16(a, make_frag((const unsigned int*)W2f2, nt * 16 + nl, 49, kt * 16, hf), d2[nt].v);
        d3[nt].v = wmma_bf16(a, make_frag((const unsigned int*)W2t,  nt * 16 + nl, 49, kt * 16, hf), d3[nt].v);
      }
    }
#pragma unroll
    for (int nt = 0; nt < 4; ++nt) {
      int n = nt * 16 + nl;
      float bv1 = b2f1[n], bv2 = b2f2[n], bvt = b2t[n];
#pragma unroll
      for (int v = 0; v < 8; ++v) {
        int m = v + hf * 8;
        float ff1 = tanhf(d1[nt].f[v] + bv1);
        float ff2 = tanhf(d2[nt].f[v] + bv2);
        float tt  = sigmf(d3[nt].f[v] + bvt);
        float nh  = ff1 * (1.0f - tt) + tt * ff2;
        hx_out[(size_t)(r0 + m) * 96 + 32 + n] = nh;   // raw nh2 -> hx_new
        lns[m * 65 + n] = nh;
      }
    }
    MEMFENCE();

    // ---- final LN on nh2 -> out ----
    if (lane < 16) {
      float* rp = lns + lane * 65;
      float s = 0.0f, s2 = 0.0f;
      for (int c = 0; c < 64; ++c) { float x = rp[c]; s += x; s2 += x * x; }
      float mu = s * 0.015625f;
      float rs = rsqrtf(s2 * 0.015625f - mu * mu + 1e-5f);
      for (int c = 0; c < 64; ++c)
        out[(size_t)(r0 + lane) * 64 + c] = (rp[c] - mu) * rs * G2[c] + B2v[c];
    }
    MEMFENCE();
  }
}

extern "C" void kernel_launch(void* const* d_in, const int* in_sizes, int n_in,
                              void* d_out, int out_size, void* d_ws, size_t ws_size,
                              hipStream_t stream) {
  (void)n_in; (void)out_size; (void)d_ws; (void)ws_size;
  const float* per_tf   = (const float*)d_in[0];
  const float* shared_f = (const float*)d_in[1];
  const float* hx       = (const float*)d_in[2];
  const float* in_w     = (const float*)d_in[3];
  const float* in_b     = (const float*)d_in[4];
  const float* ln1g     = (const float*)d_in[5];
  const float* ln1b     = (const float*)d_in[6];
  const float* ln2g     = (const float*)d_in[7];
  const float* ln2b     = (const float*)d_in[8];
  const float* f1w0 = (const float*)d_in[9];  const float* f1b0 = (const float*)d_in[10];
  const float* f2w0 = (const float*)d_in[11]; const float* f2b0 = (const float*)d_in[12];
  const float* taw0 = (const float*)d_in[13]; const float* tab0 = (const float*)d_in[14];
  const float* tbw0 = (const float*)d_in[15]; const float* tbb0 = (const float*)d_in[16];
  const float* f1w1 = (const float*)d_in[17]; const float* f1b1 = (const float*)d_in[18];
  const float* f2w1 = (const float*)d_in[19]; const float* f2b1 = (const float*)d_in[20];
  const float* taw1 = (const float*)d_in[21]; const float* tab1 = (const float*)d_in[22];
  const float* tbw1 = (const float*)d_in[23]; const float* tbb1 = (const float*)d_in[24];
  const float* f1w2 = (const float*)d_in[25]; const float* f1b2 = (const float*)d_in[26];
  const float* f2w2 = (const float*)d_in[27]; const float* f2b2 = (const float*)d_in[28];
  const float* taw2 = (const float*)d_in[29]; const float* tab2 = (const float*)d_in[30];
  const float* tbw2 = (const float*)d_in[31]; const float* tbb2 = (const float*)d_in[32];

  const int B = in_sizes[0] / 51;       // per_tf_features is [B, 51]
  const int ntiles = B / 16;
  float* out    = (float*)d_out;
  float* hx_out = out + (size_t)B * 64; // tuple output: [out | hx_new]

  int blocks = 1024;                    // 4 waves per block
  if (blocks * 4 > ntiles) blocks = (ntiles + 3) / 4;
  if (blocks < 1) blocks = 1;

  cfc_fused_kernel<<<blocks, 128, 0, stream>>>(
      per_tf, shared_f, hx, in_w, in_b, ln1g, ln1b, ln2g, ln2b,
      f1w0, f1b0, f2w0, f2b0, taw0, tab0, tbw0, tbb0,
      f1w1, f1b1, f2w1, f2b1, taw1, tab1, tbw1, tbb1,
      f1w2, f1b2, f2w2, f2b2, taw2, tab2, tbw2, tbb2,
      out, hx_out, ntiles);
}
